// SlowSelectiveScan1D_30348238913546
// MI455X (gfx1250) — matrix operand order repaired and checked by reading, not compiled, not run on hardware
//
#include <hip/hip_runtime.h>
#include <hip/hip_bf16.h>
#include <math.h>

// -------------------------------------------------------------------------
// Mamba-style selective scan, fused for MI455X (gfx1250, wave32).
//  * GEMMs: v_wmma_f32_16x16x32_f16, 64x64 block tiles, operands staged into
//    LDS by the Tensor Data Mover (tensor_load_to_lds, TENSORcnt), double-
//    buffered so DMA overlaps the matrix pipe.  TDM pad_enable pre-pads LDS
//    rows (64+8 f16) so fragment reads are bank-conflict-free.
//  * Scan: register-resident recurrence h[48] per channel (avoids the 200 MB
//    h tensor of the reference formulation).
// Workspace: ~26 MB (entirely L2-resident on the 192 MB L2).
// -------------------------------------------------------------------------

typedef __attribute__((ext_vector_type(16))) _Float16 v16h;
typedef __attribute__((ext_vector_type(8)))  _Float16 v8h;
typedef __attribute__((ext_vector_type(8)))  float    v8f;
typedef __attribute__((ext_vector_type(4)))  unsigned int u32x4;
typedef __attribute__((ext_vector_type(8)))  unsigned int u32x8;

#define DIMD    512
#define NSTATE  48
#define DTRANK  24
#define BATCH   2
#define SEQ     1024
#define ROWS    (BATCH * SEQ)   // 2048
#define PROJ    (2 * DIMD)      // 1024
#define PW      128             // padded params width (dt 24 | B 48 | C 48 | pad)
#define KDT     32              // padded dt-rank -> single WMMA K step

#define KC      64              // K chunk staged in LDS per TDM transfer
#define LDSROW  72              // f16 per LDS row: 64 data + 8 pad (16B)

// ---------------- helpers -------------------------------------------------
__device__ inline unsigned lds_off(const void* p) {
    // generic 'shared' pointers carry the LDS byte offset in their low 32 bits
    return (unsigned)(unsigned long long)(p);
}

// Issue one 2-D TDM tile load: global (row-major, rows of `stride0` f16
// elements) -> LDS, tile KC x 64 rows, 2-byte elements, row padding 16B/128B.
__device__ inline void tdm_load_tile(unsigned lds_addr, const void* gaddr,
                                     unsigned tensor_d0, unsigned long long stride0)
{
    unsigned long long ga = (unsigned long long)(size_t)gaddr;
    u32x4 g0;
    g0[0] = 1u;                                            // count=1 (user desc)
    g0[1] = lds_addr;                                      // LDS byte address
    g0[2] = (unsigned)ga;                                  // global_addr[31:0]
    g0[3] = (unsigned)((ga >> 32) & 0x01FFFFFFu)           // global_addr[56:32]
          | 0x80000000u;                                   // type=2 ("image")
    u32x8 g1;
    g1[0] = (1u << 16)                                     // data_size=2B
          | (1u << 20)                                     // pad_enable
          | (4u << 22)                                     // pad_interval: 32 DW (128B)
          | (3u << 25);                                    // pad_amount: 4 DW (16B)
    g1[1] = (tensor_d0 & 0xFFFFu) << 16;                   // tensor_dim0[15:0]
    g1[2] = (tensor_d0 >> 16) | (0u << 16);                // dim0 hi | tensor_dim1 lo
    g1[2] |= (0x00100000u & 0xFFFFu) << 16;                // tensor_dim1 = 1<<20 (lo 16)
    g1[3] = (0x00100000u >> 16) | ((unsigned)KC << 16);    // dim1 hi | tile_dim0=KC
    g1[4] = 64u;                                           // tile_dim1=64 rows (tile_dim2=0)
    g1[5] = (unsigned)(stride0 & 0xFFFFFFFFu);             // tensor_dim0_stride lo
    g1[6] = (unsigned)((stride0 >> 32) & 0xFFFFu);         // stride hi (dim1_stride unused)
    g1[7] = 0u;
    asm volatile("tensor_load_to_lds %0, %1" :: "s"(g0), "s"(g1) : "memory");
}

__device__ inline v16h load_frag2(const _Float16* row, int o0, int o1) {
    v8h lo = *reinterpret_cast<const v8h*>(row + o0);
    v8h hi = *reinterpret_cast<const v8h*>(row + o1);
    v16h r;
#pragma unroll
    for (int j = 0; j < 8; ++j) { r[j] = lo[j]; r[j + 8] = hi[j]; }
    return r;
}

// ---------------- f32 -> f16 pack (zero pad / sub-slice) ------------------
__global__ __launch_bounds__(256)
void pack_f16(_Float16* __restrict__ dst, const float* __restrict__ src,
              int dstRows, int dstCols, int srcLd, int srcColOff,
              int validRows, int validCols)
{
    int idx = blockIdx.x * 256 + threadIdx.x;
    int total = dstRows * dstCols;
    if (idx >= total) return;
    int r = idx / dstCols, c = idx % dstCols;
    float v = 0.f;
    if (r < validRows && c < validCols)
        v = src[(size_t)r * srcLd + srcColOff + c];
    dst[idx] = (_Float16)v;
}

// ---------------- TDM-staged tiled WMMA GEMM ------------------------------
// C[MxN] = A[MxK] * W[NxK]^T.  Block = 128 thr (4 waves) -> 64x64 tile,
// each wave a 32x32 quadrant (2x2 wmma accumulators).  K staged in KC=64
// chunks: wave0 DMAs the A tile, wave1 the B tile (TENSORcnt), double-buffered.
__global__ void __launch_bounds__(128)
wmma_gemm_tdm(const _Float16* __restrict__ A, const _Float16* __restrict__ W,
              float* __restrict__ C, int K, int ldc)
{
    __shared__ _Float16 sA[2][64 * LDSROW];
    __shared__ _Float16 sB[2][64 * LDSROW];

    const int tid  = threadIdx.x;
    const int wave = tid >> 5;
    const int lane = tid & 31;
    const int half = lane >> 4;
    const int lm   = lane & 15;
    const int wr   = (wave >> 1) * 32;  // wave row offset in block tile
    const int wc   = (wave & 1) * 32;   // wave col offset

    const int m0 = blockIdx.x * 64;
    const int n0 = blockIdx.y * 64;
    const int nk = K / KC;

    v8f acc00 = {}, acc01 = {}, acc10 = {}, acc11 = {};

    // prologue: fill buffer 0
    if (wave == 0)
        tdm_load_tile(lds_off(&sA[0][0]), A + (size_t)m0 * K, (unsigned)K,
                      (unsigned long long)K);
    else if (wave == 1)
        tdm_load_tile(lds_off(&sB[0][0]), W + (size_t)n0 * K, (unsigned)K,
                      (unsigned long long)K);
    __builtin_amdgcn_s_wait_tensorcnt(0);
    __syncthreads();

    for (int kc = 0; kc < nk; ++kc) {
        const int buf = kc & 1;
        if (kc + 1 < nk) {                       // prefetch next chunk via TDM
            const int k1 = (kc + 1) * KC;
            if (wave == 0)
                tdm_load_tile(lds_off(&sA[buf ^ 1][0]), A + (size_t)m0 * K + k1,
                              (unsigned)K, (unsigned long long)K);
            else if (wave == 1)
                tdm_load_tile(lds_off(&sB[buf ^ 1][0]), W + (size_t)n0 * K + k1,
                              (unsigned)K, (unsigned long long)K);
        }

        const _Float16* tA = &sA[buf][0];
        const _Float16* tB = &sB[buf][0];
#pragma unroll
        for (int ks = 0; ks < KC; ks += 32) {
            const int ao = ks + half * 8;    // A frag: K {0..7,16..23}/{8..15,24..31}
            const int bo = ks + half * 8;    // B frag: K {0..7,16..23}/{8..15,24..31}
            v16h a0 = load_frag2(tA + (wr + lm)      * LDSROW, ao, ao + 16);
            v16h a1 = load_frag2(tA + (wr + 16 + lm) * LDSROW, ao, ao + 16);
            v16h b0 = load_frag2(tB + (wc + lm)      * LDSROW, bo, bo + 16);
            v16h b1 = load_frag2(tB + (wc + 16 + lm) * LDSROW, bo, bo + 16);
            acc00 = __builtin_amdgcn_wmma_f32_16x16x32_f16(false, a0, false, b0, (short)0, acc00, false, false);
            acc01 = __builtin_amdgcn_wmma_f32_16x16x32_f16(false, a0, false, b1, (short)0, acc01, false, false);
            acc10 = __builtin_amdgcn_wmma_f32_16x16x32_f16(false, a1, false, b0, (short)0, acc10, false, false);
            acc11 = __builtin_amdgcn_wmma_f32_16x16x32_f16(false, a1, false, b1, (short)0, acc11, false, false);
        }

        if (kc + 1 < nk) __builtin_amdgcn_s_wait_tensorcnt(0);
        __syncthreads();
    }

    // epilogue: C/D layout -> row = r + 8*half, col = lane&15
    const int cb = n0 + wc + lm;
#pragma unroll
    for (int r = 0; r < 8; ++r) {
        const int mA = m0 + wr + r + half * 8;
        const int mB = mA + 16;
        C[(size_t)mA * ldc + cb]      = acc00[r];
        C[(size_t)mA * ldc + cb + 16] = acc01[r];
        C[(size_t)mB * ldc + cb]      = acc10[r];
        C[(size_t)mB * ldc + cb + 16] = acc11[r];
    }
}

// ---------------- direct WMMA GEMM (K=32) with delta epilogue -------------
__global__ void __launch_bounds__(32)
wmma_gemm_delta(const _Float16* __restrict__ A, const _Float16* __restrict__ W,
                const float* __restrict__ bias, float* __restrict__ C,
                int K, int ldc)
{
    const int lane = threadIdx.x;
    const int half = lane >> 4;
    const int lm   = lane & 15;
    const int m0 = blockIdx.x * 16;
    const int n0 = blockIdx.y * 16;

    const _Float16* Arow = A + (size_t)(m0 + lm) * K;
    const _Float16* Wrow = W + (size_t)(n0 + lm) * K;

    v8f acc = {};
    for (int k0 = 0; k0 < K; k0 += 32) {
        v16h a = load_frag2(Arow, k0 + half * 8,  k0 + half * 8 + 16);
        v16h b = load_frag2(Wrow, k0 + half * 8,  k0 + half * 8 + 16);
        acc = __builtin_amdgcn_wmma_f32_16x16x32_f16(false, a, false, b, (short)0, acc, false, false);
    }

    const int n = n0 + lm;
#pragma unroll
    for (int r = 0; r < 8; ++r) {
        int m = m0 + r + half * 8;
        float v = acc[r] + bias[n];
        v = (v > 20.f) ? v : log1pf(__expf(v));   // softplus
        v = fminf(v, 10.f);
        C[(size_t)m * ldc + n] = v;
    }
}

// ---------------- fused selective scan ------------------------------------
__global__ __launch_bounds__(64)
void scan_kernel(const float* __restrict__ delta,   // ROWS x DIMD
                 const float* __restrict__ params,  // ROWS x PW
                 const float* __restrict__ xz,      // ROWS x PROJ (u | z)
                 const float* __restrict__ A_log,   // DIMD x NSTATE
                 const float* __restrict__ Dp,      // DIMD
                 _Float16* __restrict__ g)          // ROWS x DIMD
{
    const int b = blockIdx.y;
    const int d = blockIdx.x * 64 + threadIdx.x;
    __shared__ float sBC[96];   // [0..47]=B row, [48..95]=C row

    float Ad[NSTATE], h[NSTATE];
#pragma unroll
    for (int n = 0; n < NSTATE; ++n) {
        Ad[n] = fminf(-__expf(A_log[d * NSTATE + n]), 10000.f);
        h[n] = 0.f;
    }
    const float Dd = Dp[d];

    for (int l = 0; l < SEQ; ++l) {
        const int row = b * SEQ + l;
        for (int e = threadIdx.x; e < 96; e += 64)
            sBC[e] = params[(size_t)row * PW + DTRANK + e];
        __syncthreads();

        const float dl = delta[(size_t)row * DIMD + d];
        const float uu = xz[(size_t)row * PROJ + d];
        const float du = dl * uu;
        float ys = 0.f;
#pragma unroll
        for (int n = 0; n < NSTATE; ++n) {
            float t  = fminf(fmaxf(dl * Ad[n], -30.f), 30.f);
            float dA = __expf(t);
            h[n] = fmaf(dA, h[n], du * sBC[n]);
            ys   = fmaf(sBC[48 + n], h[n], ys);
        }
        ys = fmaf(Dd, uu, ys);

        const float z  = xz[(size_t)row * PROJ + DIMD + d];
        const float sg = 1.f / (1.f + __expf(-z));
        g[(size_t)row * DIMD + d] = (_Float16)(ys * z * sg);
        __syncthreads();
    }
}

// ---------------- nan_to_num + residual + layernorm -----------------------
__global__ __launch_bounds__(256)
void ln_kernel(const float* __restrict__ x, const float* __restrict__ y,
               const float* __restrict__ gamma, const float* __restrict__ beta,
               float* __restrict__ out)
{
    const int row = blockIdx.x;
    const int tid = threadIdx.x;
    __shared__ float ssum[256], ssq[256];

    float r[2], s = 0.f, sq = 0.f;
#pragma unroll
    for (int j = 0; j < 2; ++j) {
        int i = tid + j * 256;
        float v = y[(size_t)row * DIMD + i];
        if (isnan(v))      v = 0.f;
        else if (isinf(v)) v = (v > 0.f) ? 1.f : -1.f;
        float rv = x[(size_t)row * DIMD + i] + v;
        r[j] = rv; s += rv; sq += rv * rv;
    }
    ssum[tid] = s; ssq[tid] = sq;
    __syncthreads();
    for (int off = 128; off > 0; off >>= 1) {
        if (tid < off) { ssum[tid] += ssum[tid + off]; ssq[tid] += ssq[tid + off]; }
        __syncthreads();
    }
    const float mu  = ssum[0] * (1.f / DIMD);
    const float var = ssq[0] * (1.f / DIMD) - mu * mu;
    const float inv = rsqrtf(var + 1e-5f);
#pragma unroll
    for (int j = 0; j < 2; ++j) {
        int i = tid + j * 256;
        out[(size_t)row * DIMD + i] = (r[j] - mu) * inv * gamma[i] + beta[i];
    }
}

// ---------------- host orchestration --------------------------------------
extern "C" void kernel_launch(void* const* d_in, const int* in_sizes, int n_in,
                              void* d_out, int out_size, void* d_ws, size_t ws_size,
                              hipStream_t stream)
{
    const float* x     = (const float*)d_in[0];
    const float* W_in  = (const float*)d_in[1];
    const float* W_x   = (const float*)d_in[2];
    const float* W_dt  = (const float*)d_in[3];
    const float* b_dt  = (const float*)d_in[4];
    const float* A_log = (const float*)d_in[5];
    const float* Dp    = (const float*)d_in[6];
    const float* W_out = (const float*)d_in[7];
    const float* gamma = (const float*)d_in[8];
    const float* beta  = (const float*)d_in[9];
    float* out = (float*)d_out;

    char* ws = (char*)d_ws;
    size_t off = 0;
    auto alloc = [&](size_t bytes) -> void* {
        void* p = ws + off;
        off += (bytes + 255) & ~(size_t)255;
        return p;
    };

    _Float16* Xh     = (_Float16*)alloc((size_t)ROWS * DIMD * 2);
    _Float16* Winh   = (_Float16*)alloc((size_t)PROJ * DIMD * 2);
    _Float16* Wxh    = (_Float16*)alloc((size_t)PW   * DIMD * 2);
    _Float16* Wdth   = (_Float16*)alloc((size_t)DIMD * KDT  * 2);
    _Float16* Wouth  = (_Float16*)alloc((size_t)DIMD * DIMD * 2);
    _Float16* Uh     = (_Float16*)alloc((size_t)ROWS * DIMD * 2);
    _Float16* DTh    = (_Float16*)alloc((size_t)ROWS * KDT  * 2);
    _Float16* Gh     = (_Float16*)alloc((size_t)ROWS * DIMD * 2);
    float*    xzbuf  = (float*)   alloc((size_t)ROWS * PROJ * 4);
    float*    pbuf   = (float*)   alloc((size_t)ROWS * PW   * 4);
    float*    dbuf   = (float*)   alloc((size_t)ROWS * DIMD * 4);
    float*    ybuf   = (float*)   alloc((size_t)ROWS * DIMD * 4);

    auto pk = [&](_Float16* dst, const float* src, int dR, int dC,
                  int sLd, int sOff, int vR, int vC) {
        int total = dR * dC;
        pack_f16<<<(total + 255) / 256, 256, 0, stream>>>(dst, src, dR, dC,
                                                          sLd, sOff, vR, vC);
    };

    // stage inputs to f16 (zero-padded where needed)
    pk(Xh,    x,     ROWS, DIMD, DIMD,   0, ROWS, DIMD);
    pk(Winh,  W_in,  PROJ, DIMD, DIMD,   0, PROJ, DIMD);
    pk(Wxh,   W_x,   PW,   DIMD, DIMD,   0, DTRANK + 2 * NSTATE, DIMD);
    pk(Wdth,  W_dt,  DIMD, KDT,  DTRANK, 0, DIMD, DTRANK);
    pk(Wouth, W_out, DIMD, DIMD, DIMD,   0, DIMD, DIMD);

    // 1) xz = X @ W_in^T      (2048 x 1024, K=512)  [TDM-staged WMMA]
    wmma_gemm_tdm<<<dim3(ROWS / 64, PROJ / 64), 128, 0, stream>>>(
        Xh, Winh, xzbuf, DIMD, PROJ);

    // 2) params = U @ W_x^T   (2048 x 128pad, K=512)
    pk(Uh, xzbuf, ROWS, DIMD, PROJ, 0, ROWS, DIMD);
    wmma_gemm_tdm<<<dim3(ROWS / 64, PW / 64), 128, 0, stream>>>(
        Uh, Wxh, pbuf, DIMD, PW);

    // 3) delta = min(softplus(dt @ W_dt^T + b_dt), 10)  (K=24 padded to 32)
    pk(DTh, pbuf, ROWS, KDT, PW, 0, ROWS, DTRANK);
    wmma_gemm_delta<<<dim3(ROWS / 16, DIMD / 16), 32, 0, stream>>>(
        DTh, Wdth, b_dt, dbuf, KDT, DIMD);

    // 4) fused scan -> g = (C.h + D*u) * silu(z), emitted in f16
    scan_kernel<<<dim3(DIMD / 64, BATCH), 64, 0, stream>>>(
        dbuf, pbuf, xzbuf, A_log, Dp, Gh);

    // 5) y = g @ W_out^T      (2048 x 512, K=512)  [TDM-staged WMMA]
    wmma_gemm_tdm<<<dim3(ROWS / 64, DIMD / 64), 128, 0, stream>>>(
        Gh, Wouth, ybuf, DIMD, DIMD);

    // 6) residual + nan_to_num + layernorm
    ln_kernel<<<ROWS, 256, 0, stream>>>(x, ybuf, gamma, beta, out);
}
